// GNNAttn_76613626626624
// MI455X (gfx1250) — matrix operand order repaired and checked by reading, not compiled, not run on hardware
//
#include <hip/hip_runtime.h>
#include <hip/hip_bf16.h>
#include <math.h>

// Problem constants (match reference)
#define E_TOTAL 400000
#define FEAT    256
#define HID     256
#define NCLS    64
#define HEADS   32
#define NGRP    25000

#define ET   64    // edges per block
#define MT   128   // rows per LDS tile: 0..63 = source, 64..127 = neighbor
#define LDA  264   // padded bf16 LDS row stride (bank-conflict avoidance)

// Explicit global address space so the backend emits global_load_* / global_atomic_*
// (not flat_*), keeping LOADcnt/STOREcnt (global) and DScnt (LDS) independent.
#define GAS __attribute__((address_space(1)))

typedef __bf16 v16bf  __attribute__((ext_vector_type(16)));
typedef __bf16 bf16x8 __attribute__((ext_vector_type(8)));
typedef __bf16 bf16x4 __attribute__((ext_vector_type(4)));
typedef float  v8f    __attribute__((ext_vector_type(8)));
typedef float  f32x4  __attribute__((ext_vector_type(4)));

typedef const GAS __bf16* gw_t;   // global bf16 weights
typedef const GAS float*  gf_t;   // global fp32 data

static __device__ inline __bf16 f2bf(float f) { return (__bf16)f; }

// ---- WMMA fragment loaders --------------------------------------------------
// A (16x32 bf16, M x K) from LDS: lane 0-15 -> M=lane, K 0..15;
//                                 lane 16-31 -> M=lane-16, K 16..31.
// (The compiler register-hoists the whole per-wave A panel out of the N loop —
//  intended: A stays VGPR-resident, B streams from the L2-resident weight copy.)
static __device__ inline v16bf ld_afrag(const __bf16* X, int mrow, int k0, int lane) {
  const int m  = mrow + (lane & 15);
  const int kb = k0 + ((lane >> 4) << 4);              // +16 for upper half-wave
  bf16x8 lo = *(const bf16x8*)(X + m * LDA + kb);
  bf16x8 hi = *(const bf16x8*)(X + m * LDA + kb + 8);
  return __builtin_shufflevector(lo, hi,
      0,1,2,3,4,5,6,7,8,9,10,11,12,13,14,15);
}

// B (32x16 bf16, K x N) from global; weights stored transposed [N][256] so each
// lane reads 16 contiguous bf16 (32B): lane 0-15 -> N=lane, K=0..15;
// lane 16-31 -> N=lane-16, K=16..31.
static __device__ inline v16bf ld_bfrag(gw_t Wt, int nbase, int k0, int lane) {
  const int n  = nbase + (lane & 15);
  const int kb = k0 + ((lane >> 4) << 4);              // +16 for upper half-wave
  return *(const GAS v16bf*)(Wt + n * 256 + kb);
}

static __device__ inline v8f wmma_bf16(v16bf a, v16bf b, v8f c) {
  return __builtin_amdgcn_wmma_f32_16x16x32_bf16(
      /*neg_a=*/false, a, /*neg_b=*/false, b,
      /*c_mod=*/(short)0, c, /*reuse_a=*/false, /*reuse_b=*/false);
}

// One 256-wide layer over the 128-row tile: D = relu(S @ Wt^T), bf16 out.
// 8 waves: wave covers 4 M-tiles (M-half = wv>>2) x 4 N-tiles (N-quarter = wv&3).
static __device__ inline void gemm_relu_256(const __bf16* S, __bf16* D,
                                            gw_t Wt, int wv, int lane) {
  const int mbase0 = (wv >> 2) * 64;
  const int nt0    = (wv & 3) * 4;
  for (int nt2 = 0; nt2 < 4; ++nt2) {
    const int nbase = (nt0 + nt2) * 16;
    v8f acc[4];
#pragma unroll
    for (int mt = 0; mt < 4; ++mt) acc[mt] = (v8f){0.f,0.f,0.f,0.f,0.f,0.f,0.f,0.f};
    for (int ks = 0; ks < 8; ++ks) {
      v16bf b = ld_bfrag(Wt, nbase, ks * 32, lane);
#pragma unroll
      for (int mt = 0; mt < 4; ++mt) {
        v16bf a = ld_afrag(S, mbase0 + mt * 16, ks * 32, lane);
        acc[mt] = wmma_bf16(a, b, acc[mt]);
      }
    }
    // C/D layout: VGPR r, lanes 0-15 -> (M=r, N=lane); lanes 16-31 -> (M=8+r, N=lane-16)
    const int n  = nbase + (lane & 15);
    const int hl = (lane >> 4) << 3;
#pragma unroll
    for (int mt = 0; mt < 4; ++mt) {
      const int mo = mbase0 + mt * 16 + hl;
#pragma unroll
      for (int r = 0; r < 8; ++r) {
        float f = acc[mt][r];
        f = f > 0.f ? f : 0.f;
        D[(mo + r) * LDA + n] = f2bf(f);
      }
    }
  }
}

// ---- weight prep kernels ----------------------------------------------------
__global__ void k_conv_w01(gf_t W0, gf_t W1,
                           GAS __bf16* __restrict__ W0t, GAS __bf16* __restrict__ W1t) {
  int idx = blockIdx.x * blockDim.x + threadIdx.x;   // 0 .. 2*65536-1
  int which = idx >> 16;
  int e = idx & 65535;
  int n = e >> 8, k = e & 255;                        // dst[n][k] = src[k][n]
  if (which == 0) W0t[n * 256 + k] = f2bf(W0[k * 256 + n]);
  else            W1t[n * 256 + k] = f2bf(W1[k * 256 + n]);
}

// Folded heads: W2Qt[32][256] = (W2@Wqk)^T, W2Ct[96][256] = [W2 | W2@Wk]^T.
__global__ void k_conv_w2(gf_t W2, gf_t Wqk, gf_t Wk,
                          GAS __bf16* __restrict__ W2Qt, GAS __bf16* __restrict__ W2Ct) {
  int idx = blockIdx.x * blockDim.x + threadIdx.x;   // 0 .. 128*256-1
  int n = idx >> 8, k = idx & 255;
  if (n < 32) {
    float s = 0.f;
    for (int c = 0; c < NCLS; ++c) s += W2[k * NCLS + c] * Wqk[c * HEADS + n];
    W2Qt[n * 256 + k] = f2bf(s);
  } else {
    int m = n - 32;
    if (m < NCLS) {
      W2Ct[m * 256 + k] = f2bf(W2[k * NCLS + m]);
    } else {
      int h = m - NCLS;
      float s = 0.f;
      for (int c = 0; c < NCLS; ++c) s += W2[k * NCLS + c] * Wk[c * HEADS + h];
      W2Ct[m * 256 + k] = f2bf(s);
    }
  }
}

// ---- fused main kernel ------------------------------------------------------
__global__ __launch_bounds__(256, 1)
void k_gnn_fused(const float* src_attr_p, const float* nbr_attr_p,
                 const float* ppr_scores_p, const int* ppr_idx_p,
                 const __bf16* W0t_p, const __bf16* W1t_p,
                 const __bf16* W2Qt_p, const __bf16* W2Ct_p,
                 float* __restrict__ out) {
  // Promote all global pointers to AS(1)
  gf_t src_attr   = (gf_t)src_attr_p;
  gf_t nbr_attr   = (gf_t)nbr_attr_p;
  gf_t ppr_scores = (gf_t)ppr_scores_p;
  const GAS int* ppr_idx = (const GAS int*)ppr_idx_p;
  gw_t W0t  = (gw_t)W0t_p;
  gw_t W1t  = (gw_t)W1t_p;
  gw_t W2Qt = (gw_t)W2Qt_p;
  gw_t W2Ct = (gw_t)W2Ct_p;
  GAS float* outg = (GAS float*)out;

  __shared__ __align__(16) __bf16 Xs[MT * LDA];   // activations (bf16)
  __shared__ __align__(16) __bf16 Hs[MT * LDA];   // hidden / overlay for f32 heads
  __shared__ int   gidx[ET];
  __shared__ float pprv[ET];
  __shared__ float sc[ET];

  float* Of = (float*)Hs;             // [64 x 96] : cols 0..63 = x_nbr, 64..95 = k
  float* Qf = ((float*)Hs) + ET * 96; // [64 x 32] : q

  const int tid  = threadIdx.x;
  const int lane = tid & 31;
  const int wv   = tid >> 5;
  const long e0  = (long)blockIdx.x * ET;

  if (tid < ET) {
    gidx[tid] = ppr_idx[e0 + tid];
    pprv[tid] = ppr_scores[e0 + tid];
  }

  // Stage X tile: rows 0..63 = source_attr, rows 64..127 = neighbor_attr.
  // fp32 global load (b128) -> bf16 convert -> one packed 8B LDS store.
  for (int idx = tid; idx < MT * (FEAT / 4); idx += 256) {
    int row = idx >> 6;
    int c4  = (idx & 63) << 2;
    gf_t p = (row < ET) ? (src_attr + (e0 + row) * FEAT + c4)
                        : (nbr_attr + (e0 + row - ET) * FEAT + c4);
    f32x4 v = *(const GAS f32x4*)p;
    bf16x4 w = (bf16x4){f2bf(v.x), f2bf(v.y), f2bf(v.z), f2bf(v.w)};
    *(bf16x4*)(&Xs[row * LDA + c4]) = w;
  }

  // Warm L2 for the next block's tile while this one computes (global_prefetch_b8).
  if (blockIdx.x + 1 < gridDim.x) {
    long pe = e0 + ET + (tid >> 1);                 // 2 threads per row
    int  pc = (tid & 1) * 128;
    __builtin_prefetch((const void*)(src_attr_p + pe * FEAT + pc), 0, 1);
    __builtin_prefetch((const void*)(nbr_attr_p + pe * FEAT + pc), 0, 1);
  }
  __syncthreads();

  // Layer 1: Hs = relu(Xs @ W0)
  gemm_relu_256(Xs, Hs, W0t, wv, lane);
  __syncthreads();
  // Layer 2: Xs = relu(Hs @ W1)   (Hs becomes dead -> reused as f32 head buffer)
  gemm_relu_256(Hs, Xs, W1t, wv, lane);
  __syncthreads();

  // Layer 3 heads: q = Xs[src] @ W2Q ; [x_nbr | k] = Xs[nbr] @ [W2 | W2K]
  // 32 output tiles (8 source + 24 neighbor), 4 per wave, uniform per-wave branch.
  for (int t = wv; t < 32; t += 8) {
    const bool is_src = (t < 8);
    int mt, nbase, mbase;
    gw_t Bt;
    if (is_src) { mt = t & 3;  nbase = (t >> 2) * 16;        mbase = mt * 16;      Bt = W2Qt; }
    else        { int u = t - 8; mt = u & 3; nbase = (u >> 2) * 16; mbase = ET + mt * 16; Bt = W2Ct; }
    v8f acc = (v8f){0.f,0.f,0.f,0.f,0.f,0.f,0.f,0.f};
    for (int ks = 0; ks < 8; ++ks) {
      v16bf a = ld_afrag(Xs, mbase, ks * 32, lane);
      v16bf b = ld_bfrag(Bt, nbase, ks * 32, lane);
      acc = wmma_bf16(a, b, acc);
    }
    const int n  = nbase + (lane & 15);
    const int hl = (lane >> 4) << 3;
#pragma unroll
    for (int r = 0; r < 8; ++r) {
      int e = mt * 16 + hl + r;   // edge within block
      if (is_src) Qf[e * HEADS + n] = acc[r];
      else        Of[e * 96 + n]    = acc[r];
    }
  }
  __syncthreads();

  // Attention scale per edge: sigmoid(q . k) * ppr.
  // Rotated start index de-conflicts LDS banks (stride 32/96 floats per thread).
  if (tid < ET) {
    float d = 0.f;
    const int rot = tid & (HEADS - 1);
    for (int h = 0; h < HEADS; ++h) {
      int hh = (h + rot) & (HEADS - 1);
      d += Qf[tid * HEADS + hh] * Of[tid * 96 + NCLS + hh];
    }
    float s = 1.f / (1.f + expf(-d));
    sc[tid] = s * pprv[tid];
  }
  __syncthreads();

  // Scatter: out[g] += x_nbr * scale. No-return fp32 global atomics (STOREcnt only).
  for (int idx = tid; idx < ET * NCLS; idx += 256) {
    int e = idx >> 6, c = idx & 63;
    __hip_atomic_fetch_add(&outg[(long)gidx[e] * NCLS + c], Of[e * 96 + c] * sc[e],
                           __ATOMIC_RELAXED, __HIP_MEMORY_SCOPE_AGENT);
  }
}

// ---- host launcher ----------------------------------------------------------
extern "C" void kernel_launch(void* const* d_in, const int* in_sizes, int n_in,
                              void* d_out, int out_size, void* d_ws, size_t ws_size,
                              hipStream_t stream) {
  const float* src_attr   = (const float*)d_in[0];
  const float* nbr_attr   = (const float*)d_in[1];
  const float* ppr_scores = (const float*)d_in[2];
  const int*   ppr_idx    = (const int*)d_in[3];
  // d_in[4] = neighbor_idx (unused by the reference computation)
  const float* W0  = (const float*)d_in[5];
  const float* W1  = (const float*)d_in[6];
  const float* W2  = (const float*)d_in[7];
  const float* Wqk = (const float*)d_in[8];
  const float* Wk  = (const float*)d_in[9];

  // Workspace layout (bf16, transposed weights)
  __bf16* W0t  = (__bf16*)d_ws;          // 256*256
  __bf16* W1t  = W0t + 256 * 256;        // 256*256
  __bf16* W2Qt = W1t + 256 * 256;        // 32*256  (W2 @ Wqk)^T
  __bf16* W2Ct = W2Qt + 32 * 256;        // 96*256  [W2 | W2 @ Wk]^T

  hipMemsetAsync(d_out, 0, (size_t)out_size * sizeof(float), stream);

  k_conv_w01<<<(2 * 256 * 256) / 256, 256, 0, stream>>>(
      (gf_t)W0, (gf_t)W1, (GAS __bf16*)W0t, (GAS __bf16*)W1t);
  k_conv_w2 <<<(128 * 256) / 256, 256, 0, stream>>>(
      (gf_t)W2, (gf_t)Wqk, (gf_t)Wk, (GAS __bf16*)W2Qt, (GAS __bf16*)W2Ct);

  k_gnn_fused<<<E_TOTAL / ET, 256, 0, stream>>>(
      src_attr, nbr_attr, ppr_scores, ppr_idx, W0t, W1t, W2Qt, W2Ct,
      (float*)d_out);
}
